// FrequencyDomainAttention_54503134986449
// MI455X (gfx1250) — matrix operand order repaired and checked by reading, not compiled, not run on hardware
//
#include <hip/hip_runtime.h>
#include <hip/hip_bf16.h>
#include <cstdint>
#include <cstddef>

#define EMBED 1024
#define HEADS 16
#define HDIM  64
#define BATCH 2
#define SEQ   2048
#define BS    (BATCH*SEQ)   // 4096

typedef __attribute__((ext_vector_type(16))) __bf16 v16bf_t;
typedef __attribute__((ext_vector_type(8)))  float  v8f_t;

struct alignas(16) U4 { unsigned int x, y, z, w; };
struct Frag16 { U4 lo, hi; };   // 16 bf16 = 32 bytes
static_assert(sizeof(Frag16) == 32, "frag size");

__device__ __forceinline__ v16bf_t frag_cast(const Frag16 &f) {
  return __builtin_bit_cast(v16bf_t, f);
}

__device__ __forceinline__ unsigned short f2bf(float f) {
  unsigned int x = __float_as_uint(f);
  unsigned int r = x + 0x7FFFu + ((x >> 16) & 1u);   // round-to-nearest-even
  return (unsigned short)(r >> 16);
}

// ---------------- f32 -> bf16 conversion (4 elems / thread) ----------------
struct US4p { unsigned short a, b, c, d; };

__global__ void cvt_kernel(const float* __restrict__ in,
                           unsigned short* __restrict__ out, int n4) {
  int i = blockIdx.x * blockDim.x + threadIdx.x;
  if (i >= n4) return;
  const float4 v = ((const float4*)in)[i];
  US4p o;
  o.a = f2bf(v.x); o.b = f2bf(v.y); o.c = f2bf(v.z); o.d = f2bf(v.w);
  ((US4p*)out)[i] = o;
}

// ---------------- bf16 WMMA GEMM: C[m,n] = X[m,:] . W[n,:] + bias[n] -------
// mode 0: store bf16 -> Q  [B,H,S,D]
// mode 1: store bf16 -> Krev [B,H,S,D] with circular d-reversal
// mode 2: store bf16 -> Vt [B,H,D,S] (transposed)
// mode 3: store f32  -> out [M,N] row-major (+bias)
__global__ __launch_bounds__(256) void gemm_qkvo(
    const unsigned short* __restrict__ X,   // [4096,1024] bf16
    const unsigned short* __restrict__ W,   // [1024,1024] bf16 (row = out feature)
    const float* __restrict__ bias,         // [1024] f32
    unsigned short* __restrict__ outb,
    float* __restrict__ outf,
    int mode)
{
  const int lane = threadIdx.x & 31;
  const int wave = threadIdx.x >> 5;
  const int hi   = lane >> 4;     // 0 / 1 half of wave
  const int ll   = lane & 15;

  const int m0 = blockIdx.y * 128 + (wave >> 1) * 32;
  const int n0 = blockIdx.x * 128 + (wave & 1) * 64;

  v8f_t acc[2][4];
#pragma unroll
  for (int i = 0; i < 2; ++i)
#pragma unroll
    for (int j = 0; j < 4; ++j) acc[i][j] = {};

  for (int kc = 0; kc < EMBED / 32; ++kc) {
    const int kbA = kc * 32 + hi * 8;    // A: per-lane K = kbA..+7, kbA+16..+23
    const int kbB = kc * 32 + hi * 8;    // B: per-lane K = kbB..+7, kbB+16..+23
    Frag16 a[2], b[4];
#pragma unroll
    for (int i = 0; i < 2; ++i) {
      const unsigned short* p = X + (size_t)(m0 + i * 16 + ll) * EMBED + kbA;
      a[i].lo = *(const U4*)p;
      a[i].hi = *(const U4*)(p + 16);
    }
#pragma unroll
    for (int j = 0; j < 4; ++j) {
      const unsigned short* p = W + (size_t)(n0 + j * 16 + ll) * EMBED + kbB;
      b[j].lo = *(const U4*)p;
      b[j].hi = *(const U4*)(p + 16);
    }
    if (kc + 1 < EMBED / 32) {  // hint next k-chunk (global_prefetch_b8)
      __builtin_prefetch(X + (size_t)(m0 + ll) * EMBED + (kc + 1) * 32, 0, 1);
      __builtin_prefetch(W + (size_t)(n0 + ll) * EMBED + (kc + 1) * 32, 0, 1);
    }
#pragma unroll
    for (int i = 0; i < 2; ++i) {
      const v16bf_t av = frag_cast(a[i]);
#pragma unroll
      for (int j = 0; j < 4; ++j) {
        acc[i][j] = __builtin_amdgcn_wmma_f32_16x16x32_bf16(
            false, av, false, frag_cast(b[j]), (short)0, acc[i][j], false, false);
      }
    }
  }

#pragma unroll
  for (int i = 0; i < 2; ++i) {
#pragma unroll
    for (int j = 0; j < 4; ++j) {
      const int n = n0 + j * 16 + ll;
      const float bn = bias[n];
      const int h = n >> 6, d = n & 63;
#pragma unroll
      for (int r = 0; r < 8; ++r) {
        const int m = m0 + i * 16 + hi * 8 + r;
        const float v = acc[i][j][r] + bn;
        if (mode == 3) {
          outf[(size_t)m * EMBED + n] = v;
        } else {
          const int bb = m >> 11, s = m & (SEQ - 1);
          if (mode == 0) {
            outb[(((size_t)(bb * HEADS + h)) * SEQ + s) * HDIM + d] = f2bf(v);
          } else if (mode == 1) {
            const int dr = (HDIM - d) & (HDIM - 1);   // circular reversal
            outb[(((size_t)(bb * HEADS + h)) * SEQ + s) * HDIM + dr] = f2bf(v);
          } else {
            outb[(((size_t)(bb * HEADS + h)) * HDIM + d) * SEQ + s] = f2bf(v);
          }
        }
      }
    }
  }
}

// ---------------- flash attention: softmax(8*|Q Krev^T|) V ----------------
__global__ __launch_bounds__(256) void fattn_kernel(
    const unsigned short* __restrict__ Qb,   // [B*H, S, D] bf16
    const unsigned short* __restrict__ Kr,   // [B*H, S, D] bf16 (d-reversed K)
    const unsigned short* __restrict__ Vt,   // [B*H, D, S] bf16
    unsigned short* __restrict__ AO)         // [B, S, E] bf16
{
  __shared__ alignas(16) unsigned short plds[8][16][136];  // per-wave P tile

  const int lane = threadIdx.x & 31;
  const int wave = threadIdx.x >> 5;
  const int hi   = lane >> 4;
  const int ll   = lane & 15;
  const int bh   = blockIdx.y;
  const int s0   = blockIdx.x * 128 + wave * 16;

  const unsigned short* Qh = Qb + (size_t)bh * SEQ * HDIM;
  const unsigned short* Kh = Kr + (size_t)bh * SEQ * HDIM;
  const unsigned short* Vh = Vt + (size_t)bh * HDIM * SEQ;

  // Q A-fragments for this wave's 16 rows (kept in registers)
  Frag16 qa[2];
#pragma unroll
  for (int kc = 0; kc < 2; ++kc) {
    const unsigned short* p = Qh + (size_t)(s0 + ll) * HDIM + kc * 32 + hi * 8;
    qa[kc].lo = *(const U4*)p;
    qa[kc].hi = *(const U4*)(p + 16);
  }

  float mrow[8], lrow[8];
  v8f_t oacc[4];
#pragma unroll
  for (int r = 0; r < 8; ++r) { mrow[r] = -3.0e38f; lrow[r] = 0.0f; }
#pragma unroll
  for (int j = 0; j < 4; ++j) oacc[j] = {};

  for (int t0 = 0; t0 < SEQ; t0 += 128) {
    // ---- S = Q * Krev^T  (16 x 128 per wave) ----
    v8f_t sc[8];
#pragma unroll
    for (int nt = 0; nt < 8; ++nt) {
      v8f_t c = {};
#pragma unroll
      for (int kc = 0; kc < 2; ++kc) {
        Frag16 bf;
        const unsigned short* p =
            Kh + (size_t)(t0 + nt * 16 + ll) * HDIM + kc * 32 + hi * 8;
        bf.lo = *(const U4*)p;
        bf.hi = *(const U4*)(p + 16);
        c = __builtin_amdgcn_wmma_f32_16x16x32_bf16(
            false, frag_cast(qa[kc]), false, frag_cast(bf), (short)0, c, false, false);
      }
      sc[nt] = c;
    }
    // scores = 8 * |qk|   (sqrt(D) * |real dot|)
#pragma unroll
    for (int nt = 0; nt < 8; ++nt)
#pragma unroll
      for (int r = 0; r < 8; ++r) sc[nt][r] = 8.0f * fabsf(sc[nt][r]);

    // ---- online softmax (row = element r within 16-lane half) ----
    float corr[8];
#pragma unroll
    for (int r = 0; r < 8; ++r) {
      float rm = sc[0][r];
#pragma unroll
      for (int nt = 1; nt < 8; ++nt) rm = fmaxf(rm, sc[nt][r]);
      rm = fmaxf(rm, __shfl_xor(rm, 1));
      rm = fmaxf(rm, __shfl_xor(rm, 2));
      rm = fmaxf(rm, __shfl_xor(rm, 4));
      rm = fmaxf(rm, __shfl_xor(rm, 8));
      const float mnew = fmaxf(mrow[r], rm);
      corr[r] = __expf(mrow[r] - mnew);
      mrow[r] = mnew;
    }
#pragma unroll
    for (int r = 0; r < 8; ++r) {
      float rs = 0.0f;
#pragma unroll
      for (int nt = 0; nt < 8; ++nt) {
        const float p = __expf(sc[nt][r] - mrow[r]);
        sc[nt][r] = p;
        rs += p;
      }
      rs += __shfl_xor(rs, 1);
      rs += __shfl_xor(rs, 2);
      rs += __shfl_xor(rs, 4);
      rs += __shfl_xor(rs, 8);
      lrow[r] = lrow[r] * corr[r] + rs;
#pragma unroll
      for (int j = 0; j < 4; ++j) oacc[j][r] *= corr[r];
    }

    // ---- repack P (C layout) -> LDS bf16 -> A layout frags ----
#pragma unroll
    for (int nt = 0; nt < 8; ++nt)
#pragma unroll
      for (int r = 0; r < 8; ++r)
        plds[wave][hi * 8 + r][nt * 16 + ll] = f2bf(sc[nt][r]);

    Frag16 pa[4];
#pragma unroll
    for (int kc = 0; kc < 4; ++kc) {
      const unsigned short* p = &plds[wave][ll][kc * 32 + hi * 8];
      pa[kc].lo = *(const U4*)p;
      pa[kc].hi = *(const U4*)(p + 16);
    }

    // ---- O += P * V  (V pre-transposed: B-frags contiguous) ----
#pragma unroll
    for (int j = 0; j < 4; ++j) {
#pragma unroll
      for (int kc = 0; kc < 4; ++kc) {
        Frag16 bf;
        const unsigned short* p =
            Vh + (size_t)(j * 16 + ll) * SEQ + t0 + kc * 32 + hi * 8;
        bf.lo = *(const U4*)p;
        bf.hi = *(const U4*)(p + 16);
        oacc[j] = __builtin_amdgcn_wmma_f32_16x16x32_bf16(
            false, frag_cast(pa[kc]), false, frag_cast(bf), (short)0, oacc[j],
            false, false);
      }
    }
  }

  // ---- epilogue: divide by l, store merged-head bf16 [B,S,E] ----
  const int b = bh >> 4, h = bh & 15;
#pragma unroll
  for (int j = 0; j < 4; ++j) {
    const int d = j * 16 + ll;
#pragma unroll
    for (int r = 0; r < 8; ++r) {
      const int s = s0 + hi * 8 + r;
      AO[((size_t)(b * SEQ + s)) * EMBED + h * HDIM + d] =
          f2bf(oacc[j][r] / lrow[r]);
    }
  }
}

// ---------------------------------------------------------------------------
extern "C" void kernel_launch(void* const* d_in, const int* in_sizes, int n_in,
                              void* d_out, int out_size, void* d_ws, size_t ws_size,
                              hipStream_t stream) {
  (void)in_sizes; (void)n_in; (void)out_size; (void)ws_size;
  const float* q  = (const float*)d_in[0];
  const float* k  = (const float*)d_in[1];
  const float* v  = (const float*)d_in[2];
  const float* Wq = (const float*)d_in[3];
  const float* bq = (const float*)d_in[4];
  const float* Wk = (const float*)d_in[5];
  const float* bk = (const float*)d_in[6];
  const float* Wv = (const float*)d_in[7];
  const float* bv = (const float*)d_in[8];
  const float* Wo = (const float*)d_in[9];
  const float* bo = (const float*)d_in[10];
  float* out = (float*)d_out;

  char* ws = (char*)d_ws;                       // 64 MiB total
  unsigned short* Xq  = (unsigned short*)(ws +        0);  // 8 MiB each
  unsigned short* Xk  = (unsigned short*)(ws +  8388608);
  unsigned short* Xv  = (unsigned short*)(ws + 16777216);
  unsigned short* Wqb = (unsigned short*)(ws + 25165824);  // 2 MiB each
  unsigned short* Wkb = (unsigned short*)(ws + 27262976);
  unsigned short* Wvb = (unsigned short*)(ws + 29360128);
  unsigned short* Wob = (unsigned short*)(ws + 31457280);
  unsigned short* Qb  = (unsigned short*)(ws + 33554432);
  unsigned short* Kr  = (unsigned short*)(ws + 41943040);
  unsigned short* Vt  = (unsigned short*)(ws + 50331648);
  unsigned short* AO  = (unsigned short*)(ws + 58720256);

  const int nBig4 = (BS * EMBED) / 4;      // 1,048,576
  const int nW4   = (EMBED * EMBED) / 4;   //   262,144
  cvt_kernel<<<dim3((nBig4 + 255) / 256), 256, 0, stream>>>(q,  Xq,  nBig4);
  cvt_kernel<<<dim3((nBig4 + 255) / 256), 256, 0, stream>>>(k,  Xk,  nBig4);
  cvt_kernel<<<dim3((nBig4 + 255) / 256), 256, 0, stream>>>(v,  Xv,  nBig4);
  cvt_kernel<<<dim3((nW4   + 255) / 256), 256, 0, stream>>>(Wq, Wqb, nW4);
  cvt_kernel<<<dim3((nW4   + 255) / 256), 256, 0, stream>>>(Wk, Wkb, nW4);
  cvt_kernel<<<dim3((nW4   + 255) / 256), 256, 0, stream>>>(Wv, Wvb, nW4);
  cvt_kernel<<<dim3((nW4   + 255) / 256), 256, 0, stream>>>(Wo, Wob, nW4);

  dim3 gg(EMBED / 128, BS / 128);  // (8, 32)
  gemm_qkvo<<<gg, 256, 0, stream>>>(Xq, Wqb, bq, Qb, nullptr, 0);  // Q
  gemm_qkvo<<<gg, 256, 0, stream>>>(Xk, Wkb, bk, Kr, nullptr, 1);  // Krev
  gemm_qkvo<<<gg, 256, 0, stream>>>(Xv, Wvb, bv, Vt, nullptr, 2);  // V^T

  fattn_kernel<<<dim3(SEQ / 128, BATCH * HEADS), 256, 0, stream>>>(Qb, Kr, Vt, AO);

  gemm_qkvo<<<gg, 256, 0, stream>>>(AO, Wob, bo, nullptr, out, 3); // out proj
}
